// Model_8658654069030
// MI455X (gfx1250) — compile-verified, matrix operand order checked by reading, not run on hardware
//
#include <hip/hip_runtime.h>
#include <hip/hip_bf16.h>

// ---------------------------------------------------------------------------
// BiLSTM-CRF forward (emissions + Viterbi decode) for MI455X / gfx1250.
// All GEMMs via v_wmma_f32_16x16x32_f16 (wave32 WMMA).
// ---------------------------------------------------------------------------

#define EDIM 100
#define EPAD 128
#define HID  256
#define GATES 1024      // 4*HID
#define BB   128
#define TT   512
#define KK   3

typedef __attribute__((ext_vector_type(16))) _Float16 v16h;
typedef __attribute__((ext_vector_type(8)))  _Float16 v8h;
typedef __attribute__((ext_vector_type(8)))  float    v8f;

// A/B fragment loader for 16x16x32 f16 WMMA.
// Per ISA layout: lane L (lo=L&15, hi=L>>4) holds, for its row/col (lo),
// halves [0..7] = K (kbase + hi*8 .. +7) and halves [8..15] = K (+16).
// Caller passes p = row_ptr + kbase + hi*8.
__device__ __forceinline__ v16h load_frag(const _Float16* p) {
  v8h lo = *(const v8h*)(p);
  v8h hi = *(const v8h*)(p + 16);
  return __builtin_shufflevector(lo, hi, 0,1,2,3,4,5,6,7,8,9,10,11,12,13,14,15);
}

__device__ __forceinline__ v8f wmma_f16(v16h a, v16h b, v8f c) {
  return __builtin_amdgcn_wmma_f32_16x16x32_f16(false, a, false, b, (short)0, c,
                                                false, false);
}

__device__ __forceinline__ float sigf(float x) {
  return 1.0f / (1.0f + __expf(-x));
}

// ---------------------------------------------------------------------------
// Prep: f32->f16 weight conversion (with K padding), fused biases, padded
// lin_W, zero the step counters.
// ---------------------------------------------------------------------------
__global__ __launch_bounds__(256) void prep_kernel(
    const float* __restrict__ w_ih_f, const float* __restrict__ w_ih_b,
    const float* __restrict__ w_hh_f, const float* __restrict__ w_hh_b,
    const float* __restrict__ b_ih_f, const float* __restrict__ b_hh_f,
    const float* __restrict__ b_ih_b, const float* __restrict__ b_hh_b,
    const float* __restrict__ lin_W,
    _Float16* __restrict__ wih16, _Float16* __restrict__ whh16,
    float* __restrict__ bias, _Float16* __restrict__ lin16,
    int* __restrict__ cnt)
{
  int idx = blockIdx.x * 256 + threadIdx.x;
  if (idx < 2 * GATES * HID) {           // whh16: (d,n,k) k<256
    int d = idx / (GATES * HID);
    int rem = idx - d * GATES * HID;
    int n = rem / HID, k = rem - n * HID;
    const float* w = d ? w_hh_b : w_hh_f;
    whh16[idx] = (_Float16)w[n * HID + k];
  }
  if (idx < 2 * GATES * EPAD) {          // wih16: (d,n,k) k<128, zero pad k>=100
    int d = idx / (GATES * EPAD);
    int rem = idx - d * GATES * EPAD;
    int n = rem / EPAD, k = rem - n * EPAD;
    const float* w = d ? w_ih_b : w_ih_f;
    wih16[idx] = (k < EDIM) ? (_Float16)w[n * EDIM + k] : (_Float16)0.0f;
  }
  if (idx < 2 * GATES) {                 // fused bias
    int d = idx >> 10, n = idx & (GATES - 1);
    bias[idx] = d ? (b_ih_b[n] + b_hh_b[n]) : (b_ih_f[n] + b_hh_f[n]);
  }
  if (idx < 16 * 2 * HID) {              // lin16: (16, 512), rows >= 3 zero
    int n = idx / (2 * HID), k = idx - n * (2 * HID);
    lin16[idx] = (n < KK) ? (_Float16)lin_W[n * (2 * HID) + k] : (_Float16)0.0f;
  }
  if (idx < 2 * TT) cnt[idx] = 0;        // per (dir, step) arrival counters
}

// ---------------------------------------------------------------------------
// Embedding gather -> xpad (T*B, 128) f16, row r = t*B + b, cols>=100 zero.
// ---------------------------------------------------------------------------
__global__ __launch_bounds__(256) void embed_kernel(
    const int* __restrict__ input, const float* __restrict__ embed_W,
    _Float16* __restrict__ xpad)
{
  size_t idx = (size_t)blockIdx.x * 256 + threadIdx.x;   // T*B*128 elements
  int e = (int)(idx & (EPAD - 1));
  int r = (int)(idx >> 7);
  int t = r >> 7, b = r & (BB - 1);
  int tok = input[b * TT + t];
  xpad[idx] = (e < EDIM) ? (_Float16)embed_W[(size_t)tok * EDIM + e]
                         : (_Float16)0.0f;
}

// ---------------------------------------------------------------------------
// Gate preactivations: xg[d][s][n][b] = (x_t . w_ih_d[n]) + b_ih + b_hh,
// t = s (fwd) or T-1-s (bwd).  Stored f16, batch-contiguous per column n so
// the recurrence can vector-load its accumulator fragments directly.
// grid = (4096 M-tiles, 4 = dir*2 x N-half), block = 256 (8 waves x 64 cols).
// ---------------------------------------------------------------------------
__global__ __launch_bounds__(256) void gates_kernel(
    const _Float16* __restrict__ xpad, const _Float16* __restrict__ wih16,
    const float* __restrict__ bias, _Float16* __restrict__ xg16)
{
  int tid = threadIdx.x;
  int L = tid & 31, wv = tid >> 5;
  int lo = L & 15, hi = L >> 4;
  int mbase = blockIdx.x * 16;
  int d = blockIdx.y >> 1;
  int nb = (blockIdx.y & 1) * 512 + wv * 64;

  v8f acc[4] = {};
  const _Float16* arow = xpad + (size_t)(mbase + lo) * EPAD + hi * 8;
  const _Float16* wrow = wih16 + ((size_t)d * GATES + nb + lo) * EPAD + hi * 8;

#pragma unroll
  for (int ks = 0; ks < 4; ++ks) {
    v16h a = load_frag(arow + ks * 32);
#pragma unroll
    for (int nt = 0; nt < 4; ++nt) {
      v16h b = load_frag(wrow + (size_t)nt * 16 * EPAD + ks * 32);
      acc[nt] = wmma_f16(a, b, acc[nt]);
    }
  }

  int t = mbase >> 7;                       // 8 M-tiles per timestep
  int s = d ? (TT - 1 - t) : t;
  int b0 = (mbase & (BB - 1)) + hi * 8;     // batch rows for this lane-half
#pragma unroll
  for (int nt = 0; nt < 4; ++nt) {
    int n = nb + nt * 16 + lo;
    float bi = bias[d * GATES + n];
    v8h hv;
#pragma unroll
    for (int v = 0; v < 8; ++v) hv[v] = (_Float16)(acc[nt][v] + bi);
    *(v8h*)(xg16 + ((size_t)(d * TT + s) * GATES + n) * BB + b0) = hv;
  }
}

// ---------------------------------------------------------------------------
// Persistent bidirectional LSTM recurrence.
// grid = 8 blocks: blockIdx.x = dir*4 + slice; block = 512 threads (16 waves).
// Each wave (jt = wv&3, mg = wv>>2) owns, for 2 M-tiles (32 batch rows) and
// 16 hidden units, the i/f/g/o WMMA tiles -> cell state c lives in VGPRs for
// all 512 steps.  Cross-WG sync per direction via atomic counter + s_sleep.
// ---------------------------------------------------------------------------
__global__ __launch_bounds__(512) void lstm_kernel(
    const _Float16* __restrict__ xg16, const _Float16* __restrict__ whh16,
    _Float16* __restrict__ hst, int* cnt)
{
  int d = blockIdx.x >> 2, w = blockIdx.x & 3;
  int tid = threadIdx.x;
  int L = tid & 31, wv = tid >> 5;
  int lo = L & 15, hi = L >> 4;
  int jt = wv & 3, mg = wv >> 2;
  int ncol = w * 64 + jt * 16 + lo;         // per-gate column for this lane

  v8f cc[2] = {};                            // register-resident cell state

  for (int s = 0; s < TT; ++s) {
    int t = d ? (TT - 1 - s) : s;

    if (s > 0) {                             // wait for full h(t_prev)
      if (tid == 0) {
        while (__hip_atomic_load(&cnt[d * TT + s - 1], __ATOMIC_ACQUIRE,
                                 __HIP_MEMORY_SCOPE_AGENT) < 4)
          __builtin_amdgcn_s_sleep(1);
      }
      __syncthreads();
    }
    if (s + 1 < TT)                          // pull next step's gate block
      __builtin_prefetch(xg16 + ((size_t)(d * TT + s + 1) * GATES) * BB, 0, 1);

    // acc init = gate preactivations (direct fragment-layout vector loads)
    v8f acc[2][4];
#pragma unroll
    for (int task = 0; task < 2; ++task) {
      int mt = mg * 2 + task;
#pragma unroll
      for (int g = 0; g < 4; ++g) {
        int n = g * HID + ncol;
        v8h x = *(const v8h*)(xg16 + ((size_t)(d * TT + s) * GATES + n) * BB +
                              mt * 16 + hi * 8);
#pragma unroll
        for (int v = 0; v < 8; ++v) acc[task][g][v] = (float)x[v];
      }
    }

    if (s > 0) {                             // + h(t_prev) @ w_hh^T
      int tp = d ? (TT - s) : (s - 1);
      const _Float16* hbase =
          hst + (size_t)tp * BB * (2 * HID) + d * HID + hi * 8;
      const _Float16* wbase =
          whh16 + ((size_t)d * GATES) * HID + hi * 8;
#pragma unroll
      for (int ks = 0; ks < 8; ++ks) {
        v16h a0 = load_frag(hbase + (size_t)(mg * 32 + lo) * (2 * HID) + ks * 32);
        v16h a1 = load_frag(hbase + (size_t)(mg * 32 + 16 + lo) * (2 * HID) + ks * 32);
#pragma unroll
        for (int g = 0; g < 4; ++g) {
          v16h b = load_frag(wbase + (size_t)(g * HID + ncol) * HID + ks * 32);
          acc[0][g] = wmma_f16(a0, b, acc[0][g]);
          acc[1][g] = wmma_f16(a1, b, acc[1][g]);
        }
      }
    }

    // elementwise LSTM cell (PyTorch gate order i,f,g,o) + h store
#pragma unroll
    for (int task = 0; task < 2; ++task) {
      int mt = mg * 2 + task;
#pragma unroll
      for (int v = 0; v < 8; ++v) {
        float iv = sigf(acc[task][0][v]);
        float fv = sigf(acc[task][1][v]);
        float gv = tanhf(acc[task][2][v]);
        float ov = sigf(acc[task][3][v]);
        float c = fv * cc[task][v] + iv * gv;
        cc[task][v] = c;
        float h = ov * tanhf(c);
        int br = mt * 16 + hi * 8 + v;
        hst[((size_t)t * BB + br) * (2 * HID) + d * HID + ncol] = (_Float16)h;
      }
    }

    __threadfence();
    __syncthreads();
    if (tid == 0)
      __hip_atomic_fetch_add(&cnt[d * TT + s], 1, __ATOMIC_RELEASE,
                             __HIP_MEMORY_SCOPE_AGENT);
  }
}

// ---------------------------------------------------------------------------
// Emissions: em[r][k] = h[r] . lin_W[k] + lin_b[k], N padded to 16.
// grid = 512, block = 256 (one M-tile per wave, K-loop of 16).
// ---------------------------------------------------------------------------
__global__ __launch_bounds__(256) void emis_kernel(
    const _Float16* __restrict__ hst, const _Float16* __restrict__ lin16,
    const float* __restrict__ lin_b, float* __restrict__ em)
{
  int tid = threadIdx.x;
  int L = tid & 31, wv = tid >> 5;
  int lo = L & 15, hi = L >> 4;
  int mt = blockIdx.x * 8 + wv;

  v8f acc = {};
  const _Float16* arow = hst + (size_t)(mt * 16 + lo) * (2 * HID) + hi * 8;
  const _Float16* brow = lin16 + (size_t)lo * (2 * HID) + hi * 8;
#pragma unroll
  for (int ks = 0; ks < 16; ++ks) {
    v16h a = load_frag(arow + ks * 32);
    v16h b = load_frag(brow + ks * 32);
    acc = wmma_f16(a, b, acc);
  }
  if (lo < KK) {
    float bb = lin_b[lo];
#pragma unroll
    for (int v = 0; v < 8; ++v) {
      int r = mt * 16 + hi * 8 + v;
      em[(size_t)r * KK + lo] = acc[v] + bb;
    }
  }
}

// ---------------------------------------------------------------------------
// Viterbi decode (torchcrf semantics): one thread per batch element, K=3.
// ---------------------------------------------------------------------------
__global__ __launch_bounds__(128) void viterbi_kernel(
    const float* __restrict__ em, const unsigned char* __restrict__ mask,
    const float* __restrict__ trans, const float* __restrict__ strans,
    const float* __restrict__ etrans, signed char* __restrict__ bp,
    int* __restrict__ out)
{
  int b = threadIdx.x;
  float sc[KK];
#pragma unroll
  for (int k = 0; k < KK; ++k) sc[k] = strans[k] + em[(size_t)b * KK + k];

  for (int t = 1; t < TT; ++t) {
    const float* e = em + ((size_t)t * BB + b) * KK;
    bool m = mask[b * TT + t] != 0;
    signed char* bpr = bp + ((size_t)t * BB + b) * KK;
    float ns[KK]; signed char id[KK];
#pragma unroll
    for (int k = 0; k < KK; ++k) {
      float best = sc[0] + trans[0 * KK + k];
      int bi = 0;
#pragma unroll
      for (int kp = 1; kp < KK; ++kp) {
        float c = sc[kp] + trans[kp * KK + k];
        if (c > best) { best = c; bi = kp; }   // strict > keeps first index
      }
      ns[k] = best + e[k];
      id[k] = (signed char)bi;
    }
#pragma unroll
    for (int k = 0; k < KK; ++k) {
      if (m) { sc[k] = ns[k]; bpr[k] = id[k]; }
      else   { bpr[k] = (signed char)k; }      // identity backpointer at pad
    }
  }

  int last = 0;
  float best = sc[0] + etrans[0];
#pragma unroll
  for (int k = 1; k < KK; ++k) {
    float c = sc[k] + etrans[k];
    if (c > best) { best = c; last = k; }
  }

  int tag = last;
  for (int t = TT - 1; t >= 1; --t) {
    out[b * TT + t] = mask[b * TT + t] ? tag : 0;
    tag = (int)bp[((size_t)t * BB + b) * KK + tag];
  }
  out[b * TT] = mask[b * TT] ? tag : 0;        // mask[:,0] is always true
}

// ---------------------------------------------------------------------------
// Host launcher
// ---------------------------------------------------------------------------
extern "C" void kernel_launch(void* const* d_in, const int* in_sizes, int n_in,
                              void* d_out, int out_size, void* d_ws,
                              size_t ws_size, hipStream_t stream) {
  const int*   input    = (const int*)d_in[0];
  const unsigned char* mask = (const unsigned char*)d_in[1];
  const float* embed_W  = (const float*)d_in[2];
  const float* w_ih_f   = (const float*)d_in[3];
  const float* w_hh_f   = (const float*)d_in[4];
  const float* b_ih_f   = (const float*)d_in[5];
  const float* b_hh_f   = (const float*)d_in[6];
  const float* w_ih_b   = (const float*)d_in[7];
  const float* w_hh_b   = (const float*)d_in[8];
  const float* b_ih_b   = (const float*)d_in[9];
  const float* b_hh_b   = (const float*)d_in[10];
  const float* lin_W    = (const float*)d_in[11];
  const float* lin_b    = (const float*)d_in[12];
  const float* trans    = (const float*)d_in[13];
  const float* strans   = (const float*)d_in[14];
  const float* etrans   = (const float*)d_in[15];
  int* out = (int*)d_out;

  char* ws = (char*)d_ws;
  size_t off = 0;
  auto carve = [&](size_t bytes) -> char* {
    char* p = ws + off;
    off = (off + bytes + 255) & ~(size_t)255;
    return p;
  };
  _Float16* xg16  = (_Float16*)carve((size_t)2 * TT * BB * GATES * 2); // 256MB
  _Float16* hst   = (_Float16*)carve((size_t)TT * BB * 2 * HID * 2);  // 64MB
  _Float16* xpad  = (_Float16*)carve((size_t)TT * BB * EPAD * 2);     // 16MB
  _Float16* wih16 = (_Float16*)carve((size_t)2 * GATES * EPAD * 2);
  _Float16* whh16 = (_Float16*)carve((size_t)2 * GATES * HID * 2);
  float*    bias  = (float*)carve((size_t)2 * GATES * 4);
  _Float16* lin16 = (_Float16*)carve((size_t)16 * 2 * HID * 2);
  float*    em    = (float*)carve((size_t)TT * BB * KK * 4);
  signed char* bp = (signed char*)carve((size_t)TT * BB * KK);
  int*      cnt   = (int*)carve((size_t)2 * TT * 4);

  // 1) weight conversion / bias fuse / counter reset
  prep_kernel<<<(2 * GATES * HID) / 256, 256, 0, stream>>>(
      w_ih_f, w_ih_b, w_hh_f, w_hh_b, b_ih_f, b_hh_f, b_ih_b, b_hh_b, lin_W,
      wih16, whh16, bias, lin16, cnt);

  // 2) embedding gather (padded f16)
  embed_kernel<<<((size_t)TT * BB * EPAD) / 256, 256, 0, stream>>>(
      input, embed_W, xpad);

  // 3) input-gate GEMM (both directions), fragment-layout f16 output
  dim3 ggrid(TT * BB / 16, 4);
  gates_kernel<<<ggrid, 256, 0, stream>>>(xpad, wih16, bias, xg16);

  // 4) persistent bidirectional LSTM recurrence (8 cooperating WGs)
  lstm_kernel<<<8, 512, 0, stream>>>(xg16, whh16, hst, cnt);

  // 5) emissions GEMM (N padded to 16)
  emis_kernel<<<TT * BB / 16 / 8, 256, 0, stream>>>(hst, lin16, lin_b, em);

  // 6) Viterbi decode + masked int32 output
  viterbi_kernel<<<1, BB, 0, stream>>>(em, mask, trans, strans, etrans, bp, out);
}
